// EV_DGCNN_CLS_51556787421322
// MI455X (gfx1250) — matrix operand order repaired and checked by reading, not compiled, not run on hardware
//
#include <hip/hip_runtime.h>

#define SLOPE 0.2f
#define EPSV  1e-6f
#define KNN   20

typedef __attribute__((ext_vector_type(8)))  float  v8f;
typedef __attribute__((ext_vector_type(16))) __bf16 bf16x16;

// Pack two f32 -> two bf16 -> one 32-bit LDS store (4B-aligned by construction).
__device__ __forceinline__ void store_pair_bf16(__bf16* p, float a, float b) {
    union { __bf16 h[2]; unsigned u; } t;
    t.h[0] = (__bf16)a;
    t.h[1] = (__bf16)b;
    *(unsigned*)p = t.u;
}

// ---------------------------------------------------------------------------
// KNN: per query point, top-20 by negative squared distance (DGCNN style).
// ---------------------------------------------------------------------------
template <int D>
__global__ void knn_kernel(const float* __restrict__ X, int N, int* __restrict__ idx) {
    __shared__ float tile[D][64];
    __shared__ float sqs[64];

    const int n = blockIdx.x * blockDim.x + threadIdx.x;

    float q[D];
    float sqn = 0.f;
    if (n < N) {
#pragma unroll
        for (int d = 0; d < D; ++d) {
            q[d] = X[d * N + n];
            sqn += q[d] * q[d];
        }
    }

    float bv[KNN];
    int   bi[KNN];
#pragma unroll
    for (int i = 0; i < KNN; ++i) { bv[i] = -3.4e38f; bi[i] = 0; }

    for (int base = 0; base < N; base += 64) {
        __syncthreads();
        for (int t = threadIdx.x; t < D * 64; t += blockDim.x) {
            int d = t >> 6;
            int j = t & 63;
            tile[d][j] = X[d * N + base + j];
        }
        __syncthreads();
        if (threadIdx.x < 64) {
            float s = 0.f;
#pragma unroll
            for (int d = 0; d < D; ++d) {
                float v = tile[d][threadIdx.x];
                s += v * v;
            }
            sqs[threadIdx.x] = s;
        }
        __syncthreads();

        if (n < N) {
            for (int j = 0; j < 64; ++j) {
                float dot = 0.f;
#pragma unroll
                for (int d = 0; d < D; ++d) dot += q[d] * tile[d][j];
                float nd = 2.f * dot - sqn - sqs[j];
                if (nd > bv[KNN - 1]) {
                    float cv = nd;
                    int   ci = base + j;
#pragma unroll
                    for (int i = 0; i < KNN; ++i) {
                        if (cv > bv[i]) {
                            float tv = bv[i]; int ti = bi[i];
                            bv[i] = cv; bi[i] = ci;
                            cv = tv; ci = ti;
                        }
                    }
                }
            }
        }
    }

    if (n < N) {
#pragma unroll
        for (int i = 0; i < KNN; ++i) idx[n * KNN + i] = bi[i];
    }
}

// ---------------------------------------------------------------------------
// graph_feature_cross: x (3,N) -> F (3 rows, 3*N*k cols).
// ---------------------------------------------------------------------------
__global__ void build_feat_cross(const float* __restrict__ x,
                                 const int* __restrict__ idx,
                                 const float* __restrict__ scale_p,
                                 float* __restrict__ F, int N) {
    int t = blockIdx.x * blockDim.x + threadIdx.x;
    if (t >= N * KNN) return;
    int n  = t / KNN;
    int kk = t - n * KNN;
    int m  = idx[n * KNN + kk];
    float sc = scale_p[0];

    float cx[3], nx[3];
#pragma unroll
    for (int d = 0; d < 3; ++d) { cx[d] = x[d * N + n]; nx[d] = x[d * N + m]; }
    float cr[3];
    cr[0] = nx[1] * cx[2] - nx[2] * cx[1];
    cr[1] = nx[2] * cx[0] - nx[0] * cx[2];
    cr[2] = nx[0] * cx[1] - nx[1] * cx[0];

    const int Ncols = 3 * N * KNN;
#pragma unroll
    for (int d3 = 0; d3 < 3; ++d3) {
        int col = (d3 * N + n) * KNN + kk;
        F[0 * Ncols + col] = sc * (nx[d3] - cx[d3]);
        F[1 * Ncols + col] = sc * cx[d3];
        F[2 * Ncols + col] = sc * cr[d3];
    }
}

// ---------------------------------------------------------------------------
// graph_feature: Xin (C,3,N) -> F (2C rows, 3*N*k cols) = [neigh-cen ; cen].
// ---------------------------------------------------------------------------
__global__ void build_feat(const float* __restrict__ Xin,
                           const int* __restrict__ idx,
                           float* __restrict__ F, int C, int N) {
    int total = C * 3 * N * KNN;
    int t = blockIdx.x * blockDim.x + threadIdx.x;
    if (t >= total) return;
    int kk = t % KNN;   int r = t / KNN;
    int n  = r % N;     r /= N;
    int d3 = r % 3;     int c = r / 3;

    int m = idx[n * KNN + kk];
    float cen = Xin[(c * 3 + d3) * N + n];
    float nei = Xin[(c * 3 + d3) * N + m];

    const int Ncols = 3 * N * KNN;
    int col = (d3 * N + n) * KNN + kk;
    F[(size_t)c * Ncols + col]       = nei - cen;
    F[(size_t)(C + c) * Ncols + col] = cen;
}

// ---------------------------------------------------------------------------
// Dual-output WMMA GEMM with LDS-staged bf16 fragments.
//   P = Wm * F, D = Um * F.   Wm/Um: (M,Kd) f32.  F: (Kd,Ncols) f32.
// Ncols % 128 == 0 guaranteed by launches (61440, 3072) -> no column guards.
// Block = 256 threads = 8 waves; wave w owns tile (row0..row0+15, col0+16w..).
// Staging has a block-uniform fast path (full K slab + full row tile): no
// exec divergence, packed v_cvt_pk_bf16 + b32 LDS stores. Fragments are read
// as contiguous ds_load_b128 pairs matching the ISA WMMA VGPR layouts:
//   A: K = (i&7) + (i>>3)*16 + half*8   (row-major W slab)
//   B: K = (i&7) + (i>>3)*16 + half*8                  (K-major F slab)
// ---------------------------------------------------------------------------
#define FSTR 40   // LDS row stride in bf16 units (80 B, 16B-aligned, bank-spread)

__global__ void gemm_dual_wmma(const float* __restrict__ Wm,
                               const float* __restrict__ Um,
                               const float* __restrict__ F,
                               float* __restrict__ P,
                               float* __restrict__ Dq,
                               int M, int Kd, int Ncols) {
    __shared__ __align__(16) __bf16 Ws[16 * FSTR];
    __shared__ __align__(16) __bf16 Us[16 * FSTR];
    __shared__ __align__(16) __bf16 Fs[128 * FSTR];

    const int tid  = threadIdx.x;
    const int lane = tid & 31;
    const int wave = tid >> 5;
    const int row0 = blockIdx.y * 16;
    const int col0 = blockIdx.x * 128;

    const int half = (lane >> 4) & 1;
    const int r    = lane & 15;
    const int lc   = wave * 16 + r;           // local column (0..127)
    const int colg = col0 + lc;               // global column (always in range)

    // staging coordinates
    const int rw = tid >> 4;                  // W/U: row 0..15
    const int wk = (tid & 15) * 2;            // W/U: k pair 0,2,...,30
    const int fc = tid & 127;                 // F: local column
    const int fkb = (tid >> 7) * 16;          // F: k half-slab base (0 or 16)

    v8f accP = {};
    v8f accD = {};

    union Frag { bf16x16 v; uint4 q[2]; };

    for (int k0 = 0; k0 < Kd; k0 += 32) {
        __syncthreads();

        const bool full = (k0 + 32 <= Kd) && (row0 + 16 <= M);  // block-uniform
        if (full) {
            // ---- fast path: no guards, packed stores ----
            const float* wp = &Wm[(size_t)(row0 + rw) * Kd + k0 + wk];
            const float* up = &Um[(size_t)(row0 + rw) * Kd + k0 + wk];
            store_pair_bf16(&Ws[rw * FSTR + wk], wp[0], wp[1]);
            store_pair_bf16(&Us[rw * FSTR + wk], up[0], up[1]);

            const float* fp = &F[(size_t)(k0 + fkb) * Ncols + col0 + fc];
#pragma unroll
            for (int j = 0; j < 8; ++j) {
                float f0 = fp[(size_t)(2 * j) * Ncols];
                float f1 = fp[(size_t)(2 * j + 1) * Ncols];
                store_pair_bf16(&Fs[fc * FSTR + fkb + 2 * j], f0, f1);
            }
        } else {
            // ---- tail path: zero-padded ----
            {
                int gr = row0 + rw;
                float w0 = 0.f, w1 = 0.f, u0 = 0.f, u1 = 0.f;
                if (gr < M) {
                    if (k0 + wk < Kd) {
                        w0 = Wm[(size_t)gr * Kd + k0 + wk];
                        u0 = Um[(size_t)gr * Kd + k0 + wk];
                    }
                    if (k0 + wk + 1 < Kd) {
                        w1 = Wm[(size_t)gr * Kd + k0 + wk + 1];
                        u1 = Um[(size_t)gr * Kd + k0 + wk + 1];
                    }
                }
                store_pair_bf16(&Ws[rw * FSTR + wk], w0, w1);
                store_pair_bf16(&Us[rw * FSTR + wk], u0, u1);
            }
            {
                int gc = col0 + fc;
#pragma unroll
                for (int j = 0; j < 8; ++j) {
                    int kk0 = fkb + 2 * j;
                    float f0 = 0.f, f1 = 0.f;
                    if (k0 + kk0 < Kd)     f0 = F[(size_t)(k0 + kk0) * Ncols + gc];
                    if (k0 + kk0 + 1 < Kd) f1 = F[(size_t)(k0 + kk0 + 1) * Ncols + gc];
                    store_pair_bf16(&Fs[fc * FSTR + kk0], f0, f1);
                }
            }
        }
        __syncthreads();

        if (k0 + 32 < Kd) {
            __builtin_prefetch(&F[(size_t)(k0 + 32) * Ncols + colg], 0, 1);
        }

        // ---- build fragments (contiguous 16B LDS reads) ----
        Frag fa, fu, fb;
        fa.q[0] = *(const uint4*)&Ws[r * FSTR + half * 8];
        fa.q[1] = *(const uint4*)&Ws[r * FSTR + 16 + half * 8];
        fu.q[0] = *(const uint4*)&Us[r * FSTR + half * 8];
        fu.q[1] = *(const uint4*)&Us[r * FSTR + 16 + half * 8];
        fb.q[0] = *(const uint4*)&Fs[lc * FSTR + half * 8];
        fb.q[1] = *(const uint4*)&Fs[lc * FSTR + 16 + half * 8];

        accP = __builtin_amdgcn_wmma_f32_16x16x32_bf16(false, fa.v, false, fb.v,
                                                       (short)0, accP, false, false);
        accD = __builtin_amdgcn_wmma_f32_16x16x32_bf16(false, fu.v, false, fb.v,
                                                       (short)0, accD, false, false);
    }

#pragma unroll
    for (int rr = 0; rr < 8; ++rr) {
        int row = row0 + rr + half * 8;
        if (row < M) {
            P [(size_t)row * Ncols + colg] = accP[rr];
            Dq[(size_t)row * Ncols + colg] = accD[rr];
        }
    }
}

// ---------------------------------------------------------------------------
// Vector-neuron LeakyReLU epilogue + mean over k.
// ---------------------------------------------------------------------------
__global__ void ev_epilogue(const float* __restrict__ P,
                            const float* __restrict__ Dq,
                            float* __restrict__ Out,
                            int M, int N, int k) {
    int t = blockIdx.x * blockDim.x + threadIdx.x;
    if (t >= M * N) return;
    int n = t % N;
    int m = t / N;
    size_t Ncols = (size_t)3 * N * k;

    float acc0 = 0.f, acc1 = 0.f, acc2 = 0.f;
    for (int kk = 0; kk < k; ++kk) {
        float pv[3], dv[3];
        float dot = 0.f, dsq = 0.f;
#pragma unroll
        for (int d3 = 0; d3 < 3; ++d3) {
            size_t col = ((size_t)d3 * N + n) * k + kk;
            pv[d3] = P [(size_t)m * Ncols + col];
            dv[d3] = Dq[(size_t)m * Ncols + col];
            dot += pv[d3] * dv[d3];
            dsq += dv[d3] * dv[d3];
        }
        float s = dot / (dsq + EPSV);
        float o[3];
#pragma unroll
        for (int d3 = 0; d3 < 3; ++d3) {
            float neg = pv[d3] - s * dv[d3];
            float act = (dot >= 0.f) ? pv[d3] : neg;
            o[d3] = SLOPE * pv[d3] + (1.f - SLOPE) * act;
        }
        acc0 += o[0]; acc1 += o[1]; acc2 += o[2];
    }
    float inv_k = 1.f / (float)k;
    Out[((size_t)m * 3 + 0) * N + n] = acc0 * inv_k;
    Out[((size_t)m * 3 + 1) * N + n] = acc1 * inv_k;
    Out[((size_t)m * 3 + 2) * N + n] = acc2 * inv_k;
}

// h682 = concat[h341, broadcast(mean_N(h341))]
__global__ void mean_concat(const float* __restrict__ h341,
                            float* __restrict__ h682, int C, int N) {
    int t = blockIdx.x * blockDim.x + threadIdx.x;
    if (t >= C * 3) return;
    const float* row = h341 + (size_t)t * N;
    float s = 0.f;
    for (int n = 0; n < N; ++n) s += row[n];
    s /= (float)N;
    float* o1 = h682 + (size_t)t * N;
    float* o2 = h682 + ((size_t)C * 3 + t) * N;
    for (int n = 0; n < N; ++n) { o1[n] = row[n]; o2[n] = s; }
}

// z0[o,d3,n] = sum_i sLin[o,i] * z170[i,d3,n]
__global__ void zlin_kernel(const float* __restrict__ sLin,
                            const float* __restrict__ z170,
                            float* __restrict__ z0, int N) {
    int t = blockIdx.x * blockDim.x + threadIdx.x;
    if (t >= 9 * N) return;
    int n  = t % N;
    int d3 = (t / N) % 3;
    int o  = t / (3 * N);
    float s = 0.f;
    for (int i = 0; i < 170; ++i)
        s += sLin[o * 170 + i] * z170[((size_t)i * 3 + d3) * N + n];
    z0[((size_t)o * 3 + d3) * N + n] = s;
}

// hs[i,kd,n] = sum_j h682[i,j,n] * z0[j,kd,n]
__global__ void hs_kernel(const float* __restrict__ h682,
                          const float* __restrict__ z0,
                          float* __restrict__ hs, int C, int N) {
    int t = blockIdx.x * blockDim.x + threadIdx.x;
    if (t >= C * 3 * N) return;
    int n  = t % N;
    int kd = (t / N) % 3;
    int i  = t / (3 * N);
    float s = 0.f;
#pragma unroll
    for (int j = 0; j < 3; ++j)
        s += h682[((size_t)i * 3 + j) * N + n] * z0[((size_t)j * 3 + kd) * N + n];
    hs[((size_t)i * 3 + kd) * N + n] = s;
}

// pooled[c] = max_n hs[c,n]; pooled[C2+c] = mean_n hs[c,n]
__global__ void pool_kernel(const float* __restrict__ hs,
                            float* __restrict__ pooled, int C2, int N) {
    int c = blockIdx.x * blockDim.x + threadIdx.x;
    if (c >= C2) return;
    const float* row = hs + (size_t)c * N;
    float mx = -3.4e38f, sm = 0.f;
    for (int n = 0; n < N; ++n) {
        float v = row[n];
        mx = fmaxf(mx, v);
        sm += v;
    }
    pooled[c]      = mx;
    pooled[C2 + c] = sm / (float)N;
}

// Dense: out[b,o] = act( (in[b,:]·W[o,:] + bias[o]) * g[o] + bet[o] )
__global__ void dense_kernel(const float* __restrict__ in,
                             const float* __restrict__ W,
                             const float* __restrict__ bias,
                             const float* __restrict__ g,
                             const float* __restrict__ bet,
                             float* __restrict__ out,
                             int Bb, int M, int Kd, int act) {
    int t = blockIdx.x * blockDim.x + threadIdx.x;
    if (t >= Bb * M) return;
    int o = t % M;
    int b = t / M;
    float s = bias ? bias[o] : 0.f;
    const float* ip = in + (size_t)b * Kd;
    const float* wp = W + (size_t)o * Kd;
    for (int kk = 0; kk < Kd; ++kk) s += ip[kk] * wp[kk];
    if (g) s = s * g[o] + bet[o];
    if (act) s = (s >= 0.f) ? s : SLOPE * s;
    out[(size_t)b * M + o] = s;
}

// ---------------------------------------------------------------------------
static void launch_gemm(const float* Wm, const float* Um, const float* F,
                        float* P, float* D, int M, int Kd, int Ncols,
                        hipStream_t s) {
    // Ncols must be a multiple of 128 (61440 and 3072 both are).
    dim3 grid(Ncols / 128, (M + 15) / 16);
    gemm_dual_wmma<<<grid, 256, 0, s>>>(Wm, Um, F, P, D, M, Kd, Ncols);
}

static inline int cdiv(long long a, int b) { return (int)((a + b - 1) / b); }

extern "C" void kernel_launch(void* const* d_in, const int* in_sizes, int n_in,
                              void* d_out, int out_size, void* d_ws, size_t ws_size,
                              hipStream_t stream) {
    (void)in_sizes; (void)n_in; (void)out_size; (void)ws_size;
    const int B = 8, N = 1024;

    const float* x     = (const float*)d_in[0];
    const float* scale = (const float*)d_in[1];
    const float* W1 = (const float*)d_in[2];  const float* U1 = (const float*)d_in[3];
    const float* W2 = (const float*)d_in[4];  const float* U2 = (const float*)d_in[5];
    const float* W3 = (const float*)d_in[6];  const float* U3 = (const float*)d_in[7];
    const float* W4 = (const float*)d_in[8];  const float* U4 = (const float*)d_in[9];
    const float* W5 = (const float*)d_in[10]; const float* U5 = (const float*)d_in[11];
    const float* sW1 = (const float*)d_in[12]; const float* sU1 = (const float*)d_in[13];
    const float* sW2 = (const float*)d_in[14]; const float* sU2 = (const float*)d_in[15];
    const float* sLin = (const float*)d_in[16];
    const float* l1W = (const float*)d_in[17]; const float* l1b = (const float*)d_in[18];
    const float* bn1g = (const float*)d_in[19]; const float* bn1b = (const float*)d_in[20];
    const float* l2W = (const float*)d_in[21]; const float* l2b = (const float*)d_in[22];
    const float* bn2g = (const float*)d_in[23]; const float* bn2b = (const float*)d_in[24];
    const float* l3W = (const float*)d_in[25]; const float* l3b = (const float*)d_in[26];

    float* out = (float*)d_out;

    char* ws = (char*)d_ws;
    size_t off = 0;
    auto alloc = [&](size_t bytes) -> void* {
        void* p = ws + off;
        off = (off + bytes + 255) & ~(size_t)255;
        return p;
    };
    int*   idxb  = (int*)  alloc((size_t)N * KNN * sizeof(int));
    float* hcat  = (float*)alloc((size_t)169 * 3 * N * sizeof(float));           // x1..x4 concat
    float* Fbuf  = (float*)alloc((size_t)84 * 3 * N * KNN * sizeof(float));
    float* Pbuf  = (float*)alloc((size_t)85 * 3 * N * KNN * sizeof(float));
    float* Dbuf  = (float*)alloc((size_t)85 * 3 * N * KNN * sizeof(float));
    float* h341  = (float*)alloc((size_t)341 * 3 * N * sizeof(float));
    float* h682  = (float*)alloc((size_t)682 * 3 * N * sizeof(float));
    float* z341  = (float*)alloc((size_t)341 * 3 * N * sizeof(float));
    float* z170  = (float*)alloc((size_t)170 * 3 * N * sizeof(float));
    float* z0    = (float*)alloc((size_t)9 * N * sizeof(float));
    float* pooled = (float*)alloc((size_t)B * 4092 * sizeof(float));
    float* y1    = (float*)alloc((size_t)B * 512 * sizeof(float));
    float* y2    = (float*)alloc((size_t)B * 256 * sizeof(float));

    const int NC = 3 * N * KNN;   // 61440 columns for the k-expanded GEMMs
    float* x1 = hcat;
    float* x2 = hcat + (size_t)21 * 3 * N;
    float* x3 = hcat + (size_t)42 * 3 * N;
    float* x4 = hcat + (size_t)84 * 3 * N;

    for (int b = 0; b < B; ++b) {
        const float* xb = x + (size_t)b * 3 * N;

        // ---- stage 1: cross features, W1/U1 (21x3) ----
        knn_kernel<3><<<cdiv(N, 256), 256, 0, stream>>>(xb, N, idxb);
        build_feat_cross<<<cdiv((long long)N * KNN, 256), 256, 0, stream>>>(
            xb, idxb, scale, Fbuf, N);
        launch_gemm(W1, U1, Fbuf, Pbuf, Dbuf, 21, 3, NC, stream);
        ev_epilogue<<<cdiv(21 * N, 256), 256, 0, stream>>>(Pbuf, Dbuf, x1, 21, N, KNN);

        // ---- stage 2: W2/U2 (21x42) ----
        knn_kernel<63><<<cdiv(N, 256), 256, 0, stream>>>(x1, N, idxb);
        build_feat<<<cdiv((long long)21 * 3 * N * KNN, 256), 256, 0, stream>>>(
            x1, idxb, Fbuf, 21, N);
        launch_gemm(W2, U2, Fbuf, Pbuf, Dbuf, 21, 42, NC, stream);
        ev_epilogue<<<cdiv(21 * N, 256), 256, 0, stream>>>(Pbuf, Dbuf, x2, 21, N, KNN);

        // ---- stage 3: W3/U3 (42x42) ----
        knn_kernel<63><<<cdiv(N, 256), 256, 0, stream>>>(x2, N, idxb);
        build_feat<<<cdiv((long long)21 * 3 * N * KNN, 256), 256, 0, stream>>>(
            x2, idxb, Fbuf, 21, N);
        launch_gemm(W3, U3, Fbuf, Pbuf, Dbuf, 42, 42, NC, stream);
        ev_epilogue<<<cdiv(42 * N, 256), 256, 0, stream>>>(Pbuf, Dbuf, x3, 42, N, KNN);

        // ---- stage 4: W4/U4 (85x84) ----
        knn_kernel<126><<<cdiv(N, 256), 256, 0, stream>>>(x3, N, idxb);
        build_feat<<<cdiv((long long)42 * 3 * N * KNN, 256), 256, 0, stream>>>(
            x3, idxb, Fbuf, 42, N);
        launch_gemm(W4, U4, Fbuf, Pbuf, Dbuf, 85, 84, NC, stream);
        ev_epilogue<<<cdiv(85 * N, 256), 256, 0, stream>>>(Pbuf, Dbuf, x4, 85, N, KNN);

        // ---- W5/U5 (341x169) on hcat (F = (169, 3N) contiguous) ----
        launch_gemm(W5, U5, hcat, Pbuf, Dbuf, 341, 169, 3 * N, stream);
        ev_epilogue<<<cdiv(341 * N, 256), 256, 0, stream>>>(Pbuf, Dbuf, h341, 341, N, 1);

        // ---- global-mean concat -> h682 (682, 3N) ----
        mean_concat<<<cdiv(341 * 3, 256), 256, 0, stream>>>(h341, h682, 341, N);

        // ---- std-feature stack ----
        launch_gemm(sW1, sU1, h682, Pbuf, Dbuf, 341, 682, 3 * N, stream);
        ev_epilogue<<<cdiv(341 * N, 256), 256, 0, stream>>>(Pbuf, Dbuf, z341, 341, N, 1);
        launch_gemm(sW2, sU2, z341, Pbuf, Dbuf, 170, 341, 3 * N, stream);
        ev_epilogue<<<cdiv(170 * N, 256), 256, 0, stream>>>(Pbuf, Dbuf, z170, 170, N, 1);
        zlin_kernel<<<cdiv(9 * N, 256), 256, 0, stream>>>(sLin, z170, z0, N);

        // ---- invariant features: hs -> second output region ----
        float* hsb = out + 320 + (size_t)b * 682 * 3 * N;
        hs_kernel<<<cdiv((long long)682 * 3 * N, 256), 256, 0, stream>>>(h682, z0, hsb, 682, N);

        // ---- pooling ----
        pool_kernel<<<cdiv(2046, 256), 256, 0, stream>>>(hsb, pooled + (size_t)b * 4092, 2046, N);
    }

    // ---- MLP head ----
    dense_kernel<<<cdiv(B * 512, 256), 256, 0, stream>>>(
        pooled, l1W, l1b, bn1g, bn1b, y1, B, 512, 4092, 1);
    dense_kernel<<<cdiv(B * 256, 256), 256, 0, stream>>>(
        y1, l2W, l2b, bn2g, bn2b, y2, B, 256, 512, 1);
    dense_kernel<<<cdiv(B * 40, 256), 256, 0, stream>>>(
        y2, l3W, l3b, nullptr, nullptr, out, B, 40, 256, 0);
}
